// SlotAttention_11416023073018
// MI455X (gfx1250) — matrix operand order repaired and checked by reading, not compiled, not run on hardware
//
#include <hip/hip_runtime.h>

#define BB 64
#define NN 4096
#define SS 8
#define DD 256
#define HH 512
#define ITERS 3

typedef _Float16 half_t;
typedef __attribute__((ext_vector_type(16))) _Float16 v16h;
typedef __attribute__((ext_vector_type(8)))  _Float16 v8h;
typedef __attribute__((ext_vector_type(8)))  float    v8f;

static __device__ __forceinline__ v16h mk16(v8h lo, v8h hi) {
  v16h r;
#pragma unroll
  for (int i = 0; i < 8; ++i) { r[i] = lo[i]; r[i + 8] = hi[i]; }
  return r;
}

static __device__ __forceinline__ v8f wmma16(v16h a, v16h b, v8f c) {
  // D = A(16x32 f16) * B(32x16 f16) + C(16x16 f32)
  return __builtin_amdgcn_wmma_f32_16x16x32_f16(false, a, false, b, (short)0, c, false, false);
}

static __device__ __forceinline__ float block_sum(float v, float* red, int t) {
  red[t] = v; __syncthreads();
  for (int o = 128; o > 0; o >>= 1) {
    if (t < o) red[t] += red[t + o];
    __syncthreads();
  }
  float r = red[0]; __syncthreads();
  return r;
}

static __device__ __forceinline__ float block_max(float v, float* red, int t) {
  red[t] = v; __syncthreads();
  for (int o = 128; o > 0; o >>= 1) {
    if (t < o) red[t] = fmaxf(red[t], red[t + o]);
    __syncthreads();
  }
  float r = red[0]; __syncthreads();
  return r;
}

// ---------------- weight convert (Wk, Wv -> f16) ----------------
__global__ __launch_bounds__(256) void wconv(const float* __restrict__ wk,
                                             const float* __restrict__ wv,
                                             half_t* __restrict__ wk16,
                                             half_t* __restrict__ wv16) {
  int i = blockIdx.x * 256 + threadIdx.x;
  if (i < DD * DD) {
    wk16[i] = (half_t)wk[i];
    wv16[i] = (half_t)wv[i];
  }
}

// ---------------- slots init: mu + exp(log_sigma) * noise ----------------
__global__ __launch_bounds__(256) void init_slots(const float* __restrict__ noise,
                                                  const float* __restrict__ mu,
                                                  const float* __restrict__ lsig,
                                                  float* __restrict__ slots) {
  int i = blockIdx.x * 256 + threadIdx.x;
  int d = i & (DD - 1);
  slots[i] = mu[d] + __expf(lsig[d]) * noise[i];
}

// ---------------- fused LN(inputs) -> K (blocked f16), V (blocked f16) ----------------
// 32 rows per workgroup; 8 waves: 2 row-tiles x 4 e-quarter waves, 4 e-tiles each.
// Output layouts (both blocked on the WMMA K-dim, 64 B contiguous units):
//   kblk[b][eblk(8)][n(4096)][32]   (e = logits K-dim)
//   vblk[b][nblk(128)][e(256)][32]  (n = update K-dim)
#define KST_STRIDE 264  // halfs per staged K row (16 B aligned, 4-bank step)
#define VST_STRIDE 40   // halfs per staged V row (16 B aligned, conflict-free)

__global__ __launch_bounds__(256) void ln_kv(const float* __restrict__ x,
                                             const float* __restrict__ g,
                                             const float* __restrict__ bb,
                                             const half_t* __restrict__ wk16,
                                             const half_t* __restrict__ wv16,
                                             half_t* __restrict__ kblk,
                                             half_t* __restrict__ vblk) {
  __shared__ union {
    struct { half_t xln[32 * DD]; float psum[256]; float psq[256]; } p1;
    struct { half_t kst[32 * KST_STRIDE]; half_t vst[256 * VST_STRIDE]; } p2;
  } sm;

  const int R0 = blockIdx.x * 32;  // global row base (b*N + n)
  const int t = threadIdx.x;

  // ---- phase 1: LayerNorm into LDS (f16); 8 threads per row, 32 elems each ----
  {
    const int rloc = t >> 3;
    const int oct = t & 7;
    const float* xr = x + (size_t)(R0 + rloc) * DD + oct * 32;
    float s = 0.f, sq = 0.f;
#pragma unroll
    for (int j = 0; j < 32; j += 4) {
      float4 v4 = *(const float4*)(xr + j);
      s += v4.x + v4.y + v4.z + v4.w;
      sq += v4.x * v4.x + v4.y * v4.y + v4.z * v4.z + v4.w * v4.w;
    }
    sm.p1.psum[t] = s; sm.p1.psq[t] = sq;
    __syncthreads();
    const int b8 = t & ~7;
    float fs = 0.f, fq = 0.f;
#pragma unroll
    for (int j = 0; j < 8; ++j) { fs += sm.p1.psum[b8 + j]; fq += sm.p1.psq[b8 + j]; }
    float mean = fs * (1.0f / DD);
    float var  = fq * (1.0f / DD) - mean * mean;
    float rstd = rsqrtf(var + 1e-5f);
#pragma unroll
    for (int j = 0; j < 32; j += 4) {
      float4 v4 = *(const float4*)(xr + j);
      int d = oct * 32 + j;
      sm.p1.xln[rloc * DD + d + 0] = (half_t)((v4.x - mean) * rstd * g[d + 0] + bb[d + 0]);
      sm.p1.xln[rloc * DD + d + 1] = (half_t)((v4.y - mean) * rstd * g[d + 1] + bb[d + 1]);
      sm.p1.xln[rloc * DD + d + 2] = (half_t)((v4.z - mean) * rstd * g[d + 2] + bb[d + 2]);
      sm.p1.xln[rloc * DD + d + 3] = (half_t)((v4.w - mean) * rstd * g[d + 3] + bb[d + 3]);
    }
  }
  __syncthreads();

  // ---- phase 2: preload A fragments to registers, then WMMA into staging LDS ----
  const int w = t >> 5, lane = t & 31;
  const int rt = w >> 2;           // 16-row tile 0..1
  const int eq = w & 3;            // e-quarter 0..3 (4 e-tiles each)
  const int m = lane & 15;         // A row / B column within tile
  const int half = lane >> 4;

  v16h Areg[8];
#pragma unroll
  for (int c = 0; c < 8; ++c) {
    const half_t* ap = &sm.p1.xln[(rt * 16 + m) * DD + c * 32 + 16 * half];
    Areg[c] = mk16(*(const v8h*)ap, *(const v8h*)(ap + 8));
  }
  __syncthreads();  // xln dead; LDS reused as staging

  for (int et = 0; et < 4; ++et) {
    const int e0 = (eq * 4 + et) * 16;
    v8f ack = {}, acv = {};
#pragma unroll
    for (int c = 0; c < 8; ++c) {
      const half_t* bkp = wk16 + (size_t)(e0 + m) * DD + c * 32 + 16 * half;
      v16h Bk = mk16(*(const v8h*)bkp, *(const v8h*)(bkp + 8));
      const half_t* bvp = wv16 + (size_t)(e0 + m) * DD + c * 32 + 16 * half;
      v16h Bv = mk16(*(const v8h*)bvp, *(const v8h*)(bvp + 8));
      ack = wmma16(Areg[c], Bk, ack);
      acv = wmma16(Areg[c], Bv, acv);
    }
#pragma unroll
    for (int r = 0; r < 8; ++r) {
      int rowloc = rt * 16 + r + 8 * half;  // 0..31
      sm.p2.kst[rowloc * KST_STRIDE + e0 + m] = (half_t)ack[r];
      sm.p2.vst[(e0 + m) * VST_STRIDE + rowloc] = (half_t)acv[r];
    }
  }
  __syncthreads();

  // ---- phase 3: fully-coalesced 64 B-per-thread write-out ----
  {
    const int b = R0 / NN;
    const int nb = R0 % NN;
    // K: thread t -> (n = t&31, eblk = t>>5); dense across WG
    const int n = t & 31, eblk = t >> 5;
    const half_t* src = &sm.p2.kst[n * KST_STRIDE + eblk * 32];
    half_t* dst = kblk + (((size_t)b * 8 + eblk) * NN + nb + n) * 32;
#pragma unroll
    for (int j = 0; j < 4; ++j) ((float4*)dst)[j] = ((const float4*)src)[j];
    // V: thread t -> e = t; dense across WG
    const half_t* vsrc = &sm.p2.vst[t * VST_STRIDE];
    half_t* vdst = vblk + (((size_t)b * 128 + (nb >> 5)) * DD + t) * 32;
#pragma unroll
    for (int j = 0; j < 4; ++j) ((float4*)vdst)[j] = ((const float4*)vsrc)[j];
  }
}

// ---------------- LN(slots) -> q f16, padded to 16 rows ----------------
__global__ __launch_bounds__(256) void slots_q(const float* __restrict__ slots,
                                               const float* __restrict__ g,
                                               const float* __restrict__ bb,
                                               const float* __restrict__ Wq,
                                               half_t* __restrict__ q16) {
  __shared__ float red[256];
  __shared__ float mrow[DD];
  const int row = blockIdx.x;  // B*16
  const int b = row >> 4, s = row & 15;
  const int t = threadIdx.x;
  if (s >= SS) { q16[(size_t)row * DD + t] = (half_t)0.f; return; }
  float x = slots[((size_t)b * SS + s) * DD + t];
  float mean = block_sum(x, red, t) * (1.0f / DD);
  float var  = block_sum((x - mean) * (x - mean), red, t) * (1.0f / DD);
  float rstd = rsqrtf(var + 1e-5f);
  mrow[t] = (x - mean) * rstd * g[t] + bb[t];
  __syncthreads();
  float acc = 0.f;
  const float* wr = Wq + (size_t)t * DD;
#pragma unroll 8
  for (int d = 0; d < DD; ++d) acc += mrow[d] * wr[d];
  q16[(size_t)row * DD + t] = (half_t)(acc * 0.0625f);  // D^-0.5 = 1/16
}

// ---------------- logits = q K^T via WMMA (K blocked layout) ----------------
__global__ __launch_bounds__(256) void logits_k(const half_t* __restrict__ q16,
                                                const half_t* __restrict__ kblk,
                                                float* __restrict__ logits) {
  const int w = threadIdx.x >> 5, lane = threadIdx.x & 31;
  const int tile = blockIdx.x * 8 + w;     // B * (N/16) tiles
  const int b = tile >> 8;                 // N/16 = 256
  const int n0 = (tile & 255) << 4;
  const int m = lane & 15, half = lane >> 4;
  v8f acc = {};
#pragma unroll
  for (int c = 0; c < 8; ++c) {
    const half_t* ap = q16 + ((size_t)b * 16 + m) * DD + c * 32 + 16 * half;
    v16h A = mk16(*(const v8h*)ap, *(const v8h*)(ap + 8));
    const half_t* bp = kblk + (((size_t)b * 8 + c) * NN + n0 + m) * 32 + 16 * half;
    v16h B = mk16(*(const v8h*)bp, *(const v8h*)(bp + 8));
    acc = wmma16(A, B, acc);
  }
  if (half == 0) {
#pragma unroll
    for (int r = 0; r < 8; ++r)
      logits[((size_t)b * SS + r) * NN + n0 + m] = acc[r];
  }
}

// ---------------- softmax + eps-renorm -> attn f16 (16 padded rows) ----------------
__global__ __launch_bounds__(256) void softmax_k(const float* __restrict__ logits,
                                                 half_t* __restrict__ attn) {
  __shared__ float red[256];
  const int row = blockIdx.x;  // B*16
  const int b = row >> 4, s = row & 15;
  const int t = threadIdx.x;
  half_t* arow = attn + (size_t)row * NN;
  if (s >= SS) {
    for (int j = t; j < NN; j += 256) arow[j] = (half_t)0.f;
    return;
  }
  const float* lrow = logits + ((size_t)b * SS + s) * NN;
  float lv[16];
  float mx = -1e30f;
#pragma unroll
  for (int j = 0; j < 16; j += 4) {
    float4 v4 = *(const float4*)(lrow + t * 16 + j);
    lv[j] = v4.x; lv[j + 1] = v4.y; lv[j + 2] = v4.z; lv[j + 3] = v4.w;
    mx = fmaxf(mx, fmaxf(fmaxf(v4.x, v4.y), fmaxf(v4.z, v4.w)));
  }
  mx = block_max(mx, red, t);
  float psum = 0.f;
#pragma unroll
  for (int j = 0; j < 16; ++j) { lv[j] = __expf(lv[j] - mx); psum += lv[j]; }
  float tot = block_sum(psum, red, t);
  const float inv = 1.0f / tot;
  const float renorm = 1.0f / (1.0f + (float)NN * 1e-8f);
#pragma unroll
  for (int j = 0; j < 16; ++j)
    arow[t * 16 + j] = (half_t)((lv[j] * inv + 1e-8f) * renorm);
}

// ---------------- upd = attn V via WMMA (V blocked layout) ----------------
__global__ __launch_bounds__(256) void upd_k(const half_t* __restrict__ attn,
                                             const half_t* __restrict__ vblk,
                                             float* __restrict__ upd) {
  const int w = threadIdx.x >> 5, lane = threadIdx.x & 31;
  const int task = blockIdx.x * 8 + w;  // B*16 e-tiles
  const int b = task >> 4;
  const int e0 = (task & 15) << 4;
  const int m = lane & 15, half = lane >> 4;
  const half_t* abase = attn + ((size_t)b * 16 + m) * NN;
  const half_t* bbase = vblk + ((size_t)b * 128) * DD * 32 + (size_t)(e0 + m) * 32 + 16 * half;
  v8f acc = {};
#pragma unroll 4
  for (int n0 = 0; n0 < NN; n0 += 32) {
    const half_t* ap = abase + n0 + 16 * half;
    __builtin_prefetch(ap + 128, 0, 1);
    v16h A = mk16(*(const v8h*)ap, *(const v8h*)(ap + 8));
    const half_t* bp = bbase + (size_t)(n0 >> 5) * DD * 32;
    __builtin_prefetch(bp + DD * 32 * 4, 0, 1);
    v16h B = mk16(*(const v8h*)bp, *(const v8h*)(bp + 8));
    acc = wmma16(A, B, acc);
  }
  if (half == 0) {
#pragma unroll
    for (int r = 0; r < 8; ++r)
      upd[((size_t)b * SS + r) * DD + e0 + m] = acc[r];
  }
}

// ---------------- GRU cell + residual MLP (per slot row) ----------------
__global__ __launch_bounds__(256) void gru_mlp(const float* __restrict__ upd,
                                               float* __restrict__ slots,
                                               const float* __restrict__ W_ih,
                                               const float* __restrict__ W_hh,
                                               const float* __restrict__ b_ih,
                                               const float* __restrict__ b_hh,
                                               const float* __restrict__ g,
                                               const float* __restrict__ bb,
                                               const float* __restrict__ W1,
                                               const float* __restrict__ b1,
                                               const float* __restrict__ W2,
                                               const float* __restrict__ b2) {
  __shared__ float u[DD], h[DD], mln[DD], h1[HH], red[256];
  const int row = blockIdx.x;  // B*S
  const int t = threadIdx.x;
  u[t] = upd[(size_t)row * DD + t];
  h[t] = slots[(size_t)row * DD + t];
  __syncthreads();

  float xr = b_ih[t], xz = b_ih[DD + t], xn = b_ih[2 * DD + t];
  float hr = b_hh[t], hz = b_hh[DD + t], hn = b_hh[2 * DD + t];
  const float* wi0 = W_ih + (size_t)t * DD;
  const float* wi1 = W_ih + (size_t)(DD + t) * DD;
  const float* wi2 = W_ih + (size_t)(2 * DD + t) * DD;
  const float* wh0 = W_hh + (size_t)t * DD;
  const float* wh1 = W_hh + (size_t)(DD + t) * DD;
  const float* wh2 = W_hh + (size_t)(2 * DD + t) * DD;
#pragma unroll 4
  for (int d = 0; d < DD; ++d) {
    float ud = u[d], hd = h[d];
    xr += wi0[d] * ud; xz += wi1[d] * ud; xn += wi2[d] * ud;
    hr += wh0[d] * hd; hz += wh1[d] * hd; hn += wh2[d] * hd;
  }
  float r = 1.f / (1.f + __expf(-(xr + hr)));
  float z = 1.f / (1.f + __expf(-(xz + hz)));
  float nv = tanhf(xn + r * hn);
  float sv = (1.f - z) * nv + z * h[t];

  // LayerNorm of new slots
  float mean = block_sum(sv, red, t) * (1.0f / DD);
  float var  = block_sum((sv - mean) * (sv - mean), red, t) * (1.0f / DD);
  float rstd = rsqrtf(var + 1e-5f);
  mln[t] = (sv - mean) * rstd * g[t] + bb[t];
  __syncthreads();

  // MLP hidden
  for (int hh = t; hh < HH; hh += 256) {
    float a = b1[hh];
    const float* w1r = W1 + (size_t)hh * DD;
#pragma unroll 4
    for (int d = 0; d < DD; ++d) a += mln[d] * w1r[d];
    h1[hh] = fmaxf(a, 0.f);
  }
  __syncthreads();

  float o = sv + b2[t];
  const float* w2r = W2 + (size_t)t * HH;
#pragma unroll 4
  for (int d = 0; d < HH; ++d) o += h1[d] * w2r[d];
  slots[(size_t)row * DD + t] = o;
}

// ---------------- host ----------------
extern "C" void kernel_launch(void* const* d_in, const int* in_sizes, int n_in,
                              void* d_out, int out_size, void* d_ws, size_t ws_size,
                              hipStream_t stream) {
  const float* x      = (const float*)d_in[0];
  const float* noise  = (const float*)d_in[1];
  const float* mu     = (const float*)d_in[2];
  const float* lsig   = (const float*)d_in[3];
  const float* ln_in_g   = (const float*)d_in[4];
  const float* ln_in_b   = (const float*)d_in[5];
  const float* ln_sl_g   = (const float*)d_in[6];
  const float* ln_sl_b   = (const float*)d_in[7];
  const float* ln_mlp_g  = (const float*)d_in[8];
  const float* ln_mlp_b  = (const float*)d_in[9];
  const float* Wq     = (const float*)d_in[10];
  const float* Wk     = (const float*)d_in[11];
  const float* Wv     = (const float*)d_in[12];
  const float* W_ih   = (const float*)d_in[13];
  const float* W_hh   = (const float*)d_in[14];
  const float* b_ih   = (const float*)d_in[15];
  const float* b_hh   = (const float*)d_in[16];
  const float* W1     = (const float*)d_in[17];
  const float* b1     = (const float*)d_in[18];
  const float* W2     = (const float*)d_in[19];
  const float* b2     = (const float*)d_in[20];

  float* slots = (float*)d_out;  // slots live in the output buffer

  char* ws = (char*)d_ws;
  constexpr size_t OFF_K16   = 0;
  constexpr size_t SZ_K16    = (size_t)BB * NN * DD * 2;
  constexpr size_t OFF_VT    = OFF_K16 + SZ_K16;
  constexpr size_t SZ_VT     = (size_t)BB * DD * NN * 2;
  constexpr size_t OFF_WK16  = OFF_VT + SZ_VT;
  constexpr size_t OFF_WV16  = OFF_WK16 + (size_t)DD * DD * 2;
  constexpr size_t OFF_Q16   = OFF_WV16 + (size_t)DD * DD * 2;
  constexpr size_t OFF_LOG   = OFF_Q16 + (size_t)BB * 16 * DD * 2;
  constexpr size_t OFF_ATT   = OFF_LOG + (size_t)BB * SS * NN * 4;
  constexpr size_t OFF_UPD   = OFF_ATT + (size_t)BB * 16 * NN * 2;

  half_t* kblk  = (half_t*)(ws + OFF_K16);
  half_t* vblk  = (half_t*)(ws + OFF_VT);
  half_t* wk16  = (half_t*)(ws + OFF_WK16);
  half_t* wv16  = (half_t*)(ws + OFF_WV16);
  half_t* q16   = (half_t*)(ws + OFF_Q16);
  float*  logit = (float*)(ws + OFF_LOG);
  half_t* attn  = (half_t*)(ws + OFF_ATT);
  float*  upd   = (float*)(ws + OFF_UPD);

  wconv<<<(DD * DD + 255) / 256, 256, 0, stream>>>(Wk, Wv, wk16, wv16);
  init_slots<<<(BB * SS * DD) / 256, 256, 0, stream>>>(noise, mu, lsig, slots);
  ln_kv<<<BB * NN / 32, 256, 0, stream>>>(x, ln_in_g, ln_in_b, wk16, wv16, kblk, vblk);

  for (int it = 0; it < ITERS; ++it) {
    slots_q<<<BB * 16, 256, 0, stream>>>(slots, ln_sl_g, ln_sl_b, Wq, q16);
    logits_k<<<BB * (NN / 16) / 8, 256, 0, stream>>>(q16, kblk, logit);
    softmax_k<<<BB * 16, 256, 0, stream>>>(logit, attn);
    upd_k<<<BB * 16 / 8, 256, 0, stream>>>(attn, vblk, upd);
    gru_mlp<<<BB * SS, 256, 0, stream>>>(upd, slots, W_ih, W_hh, b_ih, b_hh,
                                         ln_mlp_g, ln_mlp_b, W1, b1, W2, b2);
  }
}
